// TransformerDecoderLayer_16174846837150
// MI455X (gfx1250) — matrix operand order repaired and checked by reading, not compiled, not run on hardware
//
#include <hip/hip_runtime.h>
#include <hip/hip_bf16.h>

// ---------------------------------------------------------------------------
// SMCA-DETR decoder layer for MI455X (gfx1250, wave32, WMMA).
// Dense GEMMs use v_wmma_f32_16x16x32_bf16 with pre-converted bf16 weights
// (single v16bf vector load per B fragment); attention softmax paths are
// flash-style VALU kernels (inner dim = 32 only). fp32 I/O.
// ---------------------------------------------------------------------------

#define L_Q    300
#define B_SZ   4
#define D_MOD  256
#define NHEADS 8
#define HD     32
#define FF_DIM 2048
#define HW_SZ  5184
#define QSCALE 0.17677669529663687f   // 1/sqrt(32)
#define TQ     10                      // queries per cross-attn wave

typedef __attribute__((ext_vector_type(16))) __bf16 v16bf;
typedef __attribute__((ext_vector_type(8)))  float  v8f;

// -------------------------- f32 -> bf16 weight conversion -------------------
__global__ void cvt_bf16_kernel(const float* __restrict__ X, __bf16* __restrict__ Y, int n)
{
  const int i = blockIdx.x * 256 + threadIdx.x;
  if (i < n) Y[i] = (__bf16)X[i];
}

// -------------------------- block reduction (256 thr) ----------------------
__device__ __forceinline__ float block_sum_256(float v, float* sbuf) {
  const int tid = threadIdx.x;
#pragma unroll
  for (int off = 16; off; off >>= 1) v += __shfl_xor(v, off);
  if ((tid & 31) == 0) sbuf[tid >> 5] = v;
  __syncthreads();
  if (tid < 8) {
    float r = sbuf[tid];
#pragma unroll
    for (int off = 4; off; off >>= 1) r += __shfl_xor(r, off);
    if (tid == 0) sbuf[0] = r;
  }
  __syncthreads();
  float out = sbuf[0];
  __syncthreads();
  return out;
}

// -------------------------- WMMA GEMM --------------------------------------
// C[M,N] = act( alpha * (A(+A2)) @ W^T + alpha*bias ) (+ R)
// A: (M,K) row-major f32; Wbf: (N,K) row-major bf16 (pre-converted).
// One wave per 16x64 tile: 4 accumulators share one A fragment per k-step.
// A frag (16x32 bf16): lane m (0-15) holds row m, K = 0..15;
//                      lane m+16 holds row m, K = 16..31.
// B frag (32x16 bf16): lane n (0-15) holds col n, K = 0..15;
//                      lane n+16 holds col n, K = 16..31  (one v16bf load).
// C frag (16x16 f32):  vgpr r: lanes 0-15 -> (M=r, N=lane);
//                      lanes 16-31 -> (M=8+r, N=lane-16).
__global__ void gemm_bf16_wmma(const float* __restrict__ A, const float* __restrict__ A2,
                               const __bf16* __restrict__ Wbf, const float* __restrict__ bias,
                               const float* __restrict__ R, float* __restrict__ C,
                               int M, int N, int K, float alpha, int do_relu)
{
  const int lane = threadIdx.x;
  const int hlf  = lane >> 4;      // which half-wave
  const int l16  = lane & 15;
  const int m0 = blockIdx.x * 16;
  const int n0 = blockIdx.y * 64;

  const float* Arow  = A + (size_t)(m0 + l16) * K;
  const float* A2row = A2 ? (A2 + (size_t)(m0 + l16) * K) : nullptr;

  v8f acc[4];
#pragma unroll
  for (int t = 0; t < 4; ++t)
#pragma unroll
    for (int i = 0; i < 8; ++i) acc[t][i] = 0.0f;

  for (int k0 = 0; k0 < K; k0 += 32) {
    v16bf a;
    const int kb = k0 + hlf * 16;
    if (A2row) {
#pragma unroll
      for (int i = 0; i < 8; ++i) a[i]     = (__bf16)(Arow[kb + i]      + A2row[kb + i]);
#pragma unroll
      for (int i = 0; i < 8; ++i) a[8 + i] = (__bf16)(Arow[kb + 8 + i] + A2row[kb + 8 + i]);
    } else {
#pragma unroll
      for (int i = 0; i < 8; ++i) a[i]     = (__bf16)Arow[kb + i];
#pragma unroll
      for (int i = 0; i < 8; ++i) a[8 + i] = (__bf16)Arow[kb + 8 + i];
    }
#pragma unroll
    for (int t = 0; t < 4; ++t) {
      // 32-byte aligned vector load of 16 bf16 (K-contiguous column slice)
      const v16bf bfr = *(const v16bf*)(Wbf + (size_t)(n0 + t * 16 + l16) * K + k0 + hlf * 16);
      acc[t] = __builtin_amdgcn_wmma_f32_16x16x32_bf16(
          false, a, false, bfr, (short)0, acc[t], false, false);
    }
  }

#pragma unroll
  for (int t = 0; t < 4; ++t) {
    const int ncol = n0 + t * 16 + l16;
    const float bv = bias ? bias[ncol] : 0.0f;
#pragma unroll
    for (int r = 0; r < 8; ++r) {
      const int mrow = m0 + hlf * 8 + r;
      float v = (acc[t][r] + bv) * alpha;
      if (do_relu) v = fmaxf(v, 0.0f);
      if (R) v += R[(size_t)mrow * N + ncol];
      C[(size_t)mrow * N + ncol] = v;
    }
  }
}

// -------------------------- LayerNorm (row of 256) --------------------------
__global__ void layernorm_kernel(const float* __restrict__ X, const float* __restrict__ w,
                                 const float* __restrict__ b, float* __restrict__ Y)
{
  __shared__ float sred[8];
  const int r = blockIdx.x, tid = threadIdx.x;
  const float x = X[(size_t)r * D_MOD + tid];
  const float mean = block_sum_256(x, sred) * (1.0f / D_MOD);
  const float d = x - mean;
  const float var = block_sum_256(d * d, sred) * (1.0f / D_MOD);
  Y[(size_t)r * D_MOD + tid] = d * rsqrtf(var + 1e-5f) * w[tid] + b[tid];
}

// -------------------------- prelude: LN(n4) + heads -------------------------
__global__ void prelude_kernel(const float* __restrict__ tgt, const float* __restrict__ qpos,
  const float* __restrict__ h_w,
  const float* __restrict__ n4w, const float* __restrict__ n4b,
  const float* __restrict__ p2w, const float* __restrict__ p2b,
  const float* __restrict__ p3w, const float* __restrict__ p3b,
  const float* __restrict__ p1w0, const float* __restrict__ p1b0,
  const float* __restrict__ p1w1, const float* __restrict__ p1b1,
  const float* __restrict__ p1w2, const float* __restrict__ p1b2,
  float* __restrict__ q_add, float* __restrict__ point, float* __restrict__ scalev,
  float* __restrict__ out_inter, float* __restrict__ out_ref)
{
  __shared__ float s_out[D_MOD], s_h0[D_MOD], s_h1[D_MOD];
  __shared__ float s_off[16], s_sc[8], s_ref[2], sred[8];
  const int r = blockIdx.x, tid = threadIdx.x;
  const int b = r & (B_SZ - 1), l = r >> 2;

  const float x = tgt[(size_t)r * D_MOD + tid] + qpos[(size_t)r * D_MOD + tid];
  q_add[(size_t)r * D_MOD + tid] = x;
  const float mean = block_sum_256(x, sred) * (1.0f / D_MOD);
  const float d = x - mean;
  const float var = block_sum_256(d * d, sred) * (1.0f / D_MOD);
  s_out[tid] = d * rsqrtf(var + 1e-5f) * n4w[tid] + n4b[tid];
  __syncthreads();

  { // point MLP layer 0
    const float* wr = p1w0 + (size_t)tid * D_MOD;
    float acc = p1b0[tid];
    for (int k = 0; k < D_MOD; ++k) acc += s_out[k] * wr[k];
    s_h0[tid] = fmaxf(acc, 0.0f);
  }
  __syncthreads();
  { // point MLP layer 1
    const float* wr = p1w1 + (size_t)tid * D_MOD;
    float acc = p1b1[tid];
    for (int k = 0; k < D_MOD; ++k) acc += s_h0[k] * wr[k];
    s_h1[tid] = fmaxf(acc, 0.0f);
  }
  __syncthreads();

  if (tid < 16) {                                   // point_sigmoid_offset
    const float* wr = p2w + (size_t)tid * D_MOD;
    float acc = p2b[tid];
    for (int k = 0; k < D_MOD; ++k) acc += s_out[k] * wr[k];
    s_off[tid] = acc;
  } else if (tid >= 32 && tid < 40) {               // dynamic scale (type2)
    const int hh = tid - 32;
    const float* wr = p3w + (size_t)hh * D_MOD;
    float acc = p3b[hh];
    for (int k = 0; k < D_MOD; ++k) acc += s_out[k] * wr[k];
    s_sc[hh] = acc * acc;
  } else if (tid >= 64 && tid < 66) {               // point MLP head -> (2,)
    const int c = tid - 64;
    const float* wr = p1w2 + (size_t)c * D_MOD;
    float acc = p1b2[c];
    for (int k = 0; k < D_MOD; ++k) acc += s_h1[k] * wr[k];
    out_inter[(size_t)r * 2 + c] = acc;
    const float sg = 1.0f / (1.0f + __expf(-acc));
    s_ref[c] = h_w[b * 2 + c] * sg * (1.0f / 32.0f);
  }
  __syncthreads();

  if (tid < 16) {
    const int hh = tid >> 1, c = tid & 1;
    out_ref[(size_t)r * 16 + tid] = s_ref[c];
    point[((size_t)l * 32 + b * 8 + hh) * 2 + c] = s_ref[c] + s_off[tid];
  }
  if (tid < 8) scalev[(size_t)l * 32 + b * 8 + tid] = s_sc[tid];
}

// -------------------------- self-attention (S=300) ---------------------------
__global__ void self_attn_kernel(const float* __restrict__ Q, const float* __restrict__ K,
                                 const float* __restrict__ V, float* __restrict__ O)
{
  __shared__ float sq[HD];
  __shared__ float sc[L_Q + 4];
  const int lane = threadIdx.x;
  const int l = blockIdx.x;
  const int bh = blockIdx.y;
  const int b = bh >> 3, h = bh & 7;
  const size_t qoff = ((size_t)l * B_SZ + b) * D_MOD + h * HD;

  sq[lane] = Q[qoff + lane];
  __syncthreads();
  for (int s = lane; s < L_Q; s += 32) {
    const float* kp = K + ((size_t)s * B_SZ + b) * D_MOD + h * HD;
    float acc = 0.0f;
#pragma unroll
    for (int dd = 0; dd < HD; ++dd) acc += sq[dd] * kp[dd];
    sc[s] = acc;
  }
  __syncthreads();
  float m = -3.4e38f;
  for (int s = lane; s < L_Q; s += 32) m = fmaxf(m, sc[s]);
#pragma unroll
  for (int off = 16; off; off >>= 1) m = fmaxf(m, __shfl_xor(m, off));
  float sum = 0.0f;
  for (int s = lane; s < L_Q; s += 32) { const float p = __expf(sc[s] - m); sc[s] = p; sum += p; }
#pragma unroll
  for (int off = 16; off; off >>= 1) sum += __shfl_xor(sum, off);
  __syncthreads();
  float o = 0.0f;
  for (int s = 0; s < L_Q; ++s)
    o += sc[s] * V[((size_t)s * B_SZ + b) * D_MOD + h * HD + lane];
  O[qoff + lane] = o / sum;
}

// -------------- gaussian-biased cross-attention (flash, S=5184) --------------
__global__ void cross_attn_kernel(const float* __restrict__ Q, const float* __restrict__ K,
                                  const float* __restrict__ V, const float* __restrict__ grid2,
                                  const float* __restrict__ point, const float* __restrict__ scalev,
                                  float* __restrict__ O)
{
  __shared__ float sq[TQ * HD];
  __shared__ float sK[32 * 33];
  __shared__ float sV[32 * 33];
  const int lane = threadIdx.x;
  const int qt = blockIdx.x;           // query tile (TQ queries)
  const int g  = blockIdx.y;           // b*8+h
  const int b = g >> 3, h = g & 7;

  for (int idx = lane; idx < TQ * HD; idx += 32) {
    const int t = idx >> 5, dd = idx & 31;
    sq[idx] = Q[((size_t)(qt * TQ + t) * B_SZ + b) * D_MOD + h * HD + dd];
  }
  float mreg[TQ], ssum[TQ], o[TQ], ptx[TQ], pty[TQ], psc[TQ];
#pragma unroll
  for (int t = 0; t < TQ; ++t) {
    mreg[t] = -3.4e38f; ssum[t] = 0.0f; o[t] = 0.0f;
    const int l = qt * TQ + t;
    ptx[t] = point[((size_t)l * 32 + g) * 2 + 0];
    pty[t] = point[((size_t)l * 32 + g) * 2 + 1];
    psc[t] = scalev[(size_t)l * 32 + g];
  }
  __syncthreads();

  for (int s0 = 0; s0 < HW_SZ; s0 += 32) {
    for (int idx = lane; idx < 32 * HD; idx += 32) {
      const int kr = idx >> 5, kc = idx & 31;
      const size_t off = ((size_t)(s0 + kr) * B_SZ + b) * D_MOD + h * HD + kc;
      sK[kr * 33 + kc] = K[off];
      sV[kr * 33 + kc] = V[off];
    }
    __syncthreads();
    const int s = s0 + lane;
    const float gx = grid2[2 * s], gy = grid2[2 * s + 1];
#pragma unroll
    for (int t = 0; t < TQ; ++t) {
      float acc = 0.0f;
#pragma unroll
      for (int dd = 0; dd < HD; ++dd) acc += sq[t * HD + dd] * sK[lane * 33 + dd];
      const float dx = ptx[t] - gx, dy = pty[t] - gy;
      const float score = acc - fabsf((dx * dx + dy * dy) * psc[t]) * 0.125f;
      float cm = score;
#pragma unroll
      for (int off = 16; off; off >>= 1) cm = fmaxf(cm, __shfl_xor(cm, off));
      const float mnew = fmaxf(mreg[t], cm);
      const float corr = __expf(mreg[t] - mnew);
      const float p = __expf(score - mnew);
      float psum = p;
#pragma unroll
      for (int off = 16; off; off >>= 1) psum += __shfl_xor(psum, off);
      ssum[t] = ssum[t] * corr + psum;
      mreg[t] = mnew;
      float ot = o[t] * corr;
      for (int j = 0; j < 32; ++j) ot += __shfl(p, j) * sV[j * 33 + lane];
      o[t] = ot;
    }
    __syncthreads();
  }
#pragma unroll
  for (int t = 0; t < TQ; ++t)
    O[((size_t)(qt * TQ + t) * B_SZ + b) * D_MOD + h * HD + lane] = o[t] / ssum[t];
}

// ---------------------------------------------------------------------------
extern "C" void kernel_launch(void* const* d_in, const int* in_sizes, int n_in,
                              void* d_out, int out_size, void* d_ws, size_t ws_size,
                              hipStream_t stream)
{
  (void)in_sizes; (void)n_in; (void)out_size; (void)ws_size;

  const float* grid_in = (const float*)d_in[0];
  const float* h_w     = (const float*)d_in[1];
  const float* tgt     = (const float*)d_in[2];
  const float* memory  = (const float*)d_in[3];
  const float* qpos    = (const float*)d_in[4];
  const float* pos     = (const float*)d_in[5];
  const float* sa_w    = (const float*)d_in[6];
  const float* sa_b    = (const float*)d_in[7];
  const float* sa_ow   = (const float*)d_in[8];
  const float* sa_ob   = (const float*)d_in[9];
  const float* ca_w    = (const float*)d_in[10];
  const float* ca_b    = (const float*)d_in[11];
  const float* ca_ow   = (const float*)d_in[12];
  const float* ca_ob   = (const float*)d_in[13];
  const float* w1      = (const float*)d_in[14];
  const float* b1      = (const float*)d_in[15];
  const float* w2      = (const float*)d_in[16];
  const float* b2      = (const float*)d_in[17];
  const float* n1w     = (const float*)d_in[18];
  const float* n1b     = (const float*)d_in[19];
  const float* n2w     = (const float*)d_in[20];
  const float* n2b     = (const float*)d_in[21];
  const float* n3w     = (const float*)d_in[22];
  const float* n3b     = (const float*)d_in[23];
  const float* n4w     = (const float*)d_in[24];
  const float* n4b     = (const float*)d_in[25];
  const float* p1w0    = (const float*)d_in[26];
  const float* p1b0    = (const float*)d_in[27];
  const float* p1w1    = (const float*)d_in[28];
  const float* p1b1    = (const float*)d_in[29];
  const float* p1w2    = (const float*)d_in[30];
  const float* p1b2    = (const float*)d_in[31];
  const float* p2w     = (const float*)d_in[32];
  const float* p2b     = (const float*)d_in[33];
  const float* p3w     = (const float*)d_in[34];
  const float* p3b     = (const float*)d_in[35];

  float* Wk = (float*)d_ws;
  const size_t RD = (size_t)L_Q * B_SZ * D_MOD;          // 307200
  float* q_add = Wk;
  float* Qs    = Wk + 1 * RD;
  float* Ks    = Wk + 2 * RD;
  float* Vs    = Wk + 3 * RD;
  float* Oself = Wk + 4 * RD;
  float* x1    = Wk + 5 * RD;
  float* tgt1  = Wk + 6 * RD;
  float* Qc    = Wk + 7 * RD;
  float* Ocr   = Wk + 8 * RD;
  float* x2    = Wk + 9 * RD;
  float* tgt2n = Wk + 10 * RD;
  float* x3    = Wk + 11 * RD;
  float* Hff   = Wk + 12 * RD;                           // L*B*FF
  float* Kc    = Hff + (size_t)L_Q * B_SZ * FF_DIM;      // HW*B*D
  float* Vc    = Kc + (size_t)HW_SZ * B_SZ * D_MOD;
  float* pnt   = Vc + (size_t)HW_SZ * B_SZ * D_MOD;      // L*32*2
  float* scl   = pnt + (size_t)L_Q * 32 * 2;             // L*32
  // bf16 weight cache (32-byte aligned: offset 16789632 floats = 67158528 B)
  __bf16* bf_base  = (__bf16*)(scl + (size_t)L_Q * 32);
  __bf16* sa_w_bf  = bf_base;                            // 768*256
  __bf16* sa_ow_bf = sa_w_bf  + (size_t)3 * D_MOD * D_MOD;
  __bf16* ca_w_bf  = sa_ow_bf + (size_t)D_MOD * D_MOD;
  __bf16* ca_ow_bf = ca_w_bf  + (size_t)3 * D_MOD * D_MOD;
  __bf16* w1_bf    = ca_ow_bf + (size_t)D_MOD * D_MOD;
  __bf16* w2_bf    = w1_bf    + (size_t)FF_DIM * D_MOD;

  float* out_tgt   = (float*)d_out;
  float* out_inter = out_tgt + RD;
  float* out_ref   = out_inter + (size_t)L_Q * B_SZ * 2;

  const int M1 = L_Q * B_SZ;      // 1200
  const int M2 = HW_SZ * B_SZ;    // 20736
  const dim3 wblk(32, 1, 1);
  const dim3 gP(M1 / 16, D_MOD / 64);
  const dim3 gK(M2 / 16, D_MOD / 64);
  const dim3 gF(M1 / 16, FF_DIM / 64);

  // weight conversion to bf16 (once per launch; ~1.57M elements total)
  {
    const int n1 = 3 * D_MOD * D_MOD, n2 = D_MOD * D_MOD, n3 = FF_DIM * D_MOD;
    cvt_bf16_kernel<<<(n1 + 255) / 256, 256, 0, stream>>>(sa_w,  sa_w_bf,  n1);
    cvt_bf16_kernel<<<(n2 + 255) / 256, 256, 0, stream>>>(sa_ow, sa_ow_bf, n2);
    cvt_bf16_kernel<<<(n1 + 255) / 256, 256, 0, stream>>>(ca_w,  ca_w_bf,  n1);
    cvt_bf16_kernel<<<(n2 + 255) / 256, 256, 0, stream>>>(ca_ow, ca_ow_bf, n2);
    cvt_bf16_kernel<<<(n3 + 255) / 256, 256, 0, stream>>>(w1,    w1_bf,    n3);
    cvt_bf16_kernel<<<(n3 + 255) / 256, 256, 0, stream>>>(w2,    w2_bf,    n3);
  }

  // prelude: q_add, LN(n4) heads, point/scale, inter/ref outputs
  prelude_kernel<<<M1, 256, 0, stream>>>(tgt, qpos, h_w, n4w, n4b, p2w, p2b, p3w, p3b,
      p1w0, p1b0, p1w1, p1b1, p1w2, p1b2, q_add, pnt, scl, out_inter, out_ref);

  // self-attention QKV projections (WMMA)
  gemm_bf16_wmma<<<gP, wblk, 0, stream>>>(q_add, nullptr, sa_w_bf,                     sa_b,             nullptr, Qs, M1, D_MOD, D_MOD, QSCALE, 0);
  gemm_bf16_wmma<<<gP, wblk, 0, stream>>>(q_add, nullptr, sa_w_bf + D_MOD * D_MOD,     sa_b + D_MOD,     nullptr, Ks, M1, D_MOD, D_MOD, 1.0f,   0);
  gemm_bf16_wmma<<<gP, wblk, 0, stream>>>(tgt,   nullptr, sa_w_bf + 2 * D_MOD * D_MOD, sa_b + 2 * D_MOD, nullptr, Vs, M1, D_MOD, D_MOD, 1.0f,   0);

  self_attn_kernel<<<dim3(L_Q, B_SZ * NHEADS), wblk, 0, stream>>>(Qs, Ks, Vs, Oself);

  // out-proj + residual, LN(n1)
  gemm_bf16_wmma<<<gP, wblk, 0, stream>>>(Oself, nullptr, sa_ow_bf, sa_ob, tgt, x1, M1, D_MOD, D_MOD, 1.0f, 0);
  layernorm_kernel<<<M1, 256, 0, stream>>>(x1, n1w, n1b, tgt1);

  // cross-attention projections (WMMA); A2 fuses +query_pos / +pos adds
  gemm_bf16_wmma<<<gP, wblk, 0, stream>>>(tgt1,   qpos,    ca_w_bf,                     ca_b,             nullptr, Qc, M1, D_MOD, D_MOD, QSCALE, 0);
  gemm_bf16_wmma<<<gK, wblk, 0, stream>>>(memory, pos,     ca_w_bf + D_MOD * D_MOD,     ca_b + D_MOD,     nullptr, Kc, M2, D_MOD, D_MOD, 1.0f,   0);
  gemm_bf16_wmma<<<gK, wblk, 0, stream>>>(memory, nullptr, ca_w_bf + 2 * D_MOD * D_MOD, ca_b + 2 * D_MOD, nullptr, Vc, M2, D_MOD, D_MOD, 1.0f,   0);

  cross_attn_kernel<<<dim3(L_Q / TQ, B_SZ * NHEADS), wblk, 0, stream>>>(Qc, Kc, Vc, grid_in, pnt, scl, Ocr);

  // cross out-proj + residual, LN(n2)
  gemm_bf16_wmma<<<gP, wblk, 0, stream>>>(Ocr, nullptr, ca_ow_bf, ca_ob, tgt1, x2, M1, D_MOD, D_MOD, 1.0f, 0);
  layernorm_kernel<<<M1, 256, 0, stream>>>(x2, n2w, n2b, tgt2n);

  // FFN (WMMA) + residual, LN(n3) straight into d_out
  gemm_bf16_wmma<<<gF, wblk, 0, stream>>>(tgt2n, nullptr, w1_bf, b1, nullptr, Hff, M1, FF_DIM, D_MOD, 1.0f, 1);
  gemm_bf16_wmma<<<gP, wblk, 0, stream>>>(Hff,   nullptr, w2_bf, b2, tgt2n,  x3,  M1, D_MOD, FF_DIM, 1.0f, 0);
  layernorm_kernel<<<M1, 256, 0, stream>>>(x3, n3w, n3b, out_tgt);
}
